// MemoryCompressedAttention_85177791414353
// MI455X (gfx1250) — matrix operand order repaired and checked by reading, not compiled, not run on hardware
//
#include <hip/hip_runtime.h>
#include <hip/hip_bf16.h>
#include <cstdint>
#include <cstddef>

// ---------------------------------------------------------------------------
// MI455X (gfx1250) Memory-Compressed Attention
//   b=2, t=2048, d=1024, h=16, dh=64, CF=3, j = 684 (incl. null), JPAD=704
//   All matrix math on V_WMMA_F32_16X16X32_BF16 (f32 accumulate).
//   GEMMs are software-pipelined two deep so WMMA overlaps global loads.
// ---------------------------------------------------------------------------

typedef __bf16 bf16_t;
typedef bf16_t v16bf __attribute__((ext_vector_type(16)));
typedef bf16_t v8bf  __attribute__((ext_vector_type(8)));
typedef float  v8f   __attribute__((ext_vector_type(8)));

union BF16x16 { v16bf v; v8bf h[2]; unsigned int u[8]; };

#define B_SZ   2
#define T_SZ   2048
#define D_SZ   1024
#define H_SZ   16
#define DH_SZ  64
#define JTOT   684          // null + 683 compressed keys
#define JPAD   704          // padded to multiple of 32
#define QSCALE 0.03125f     // 1024^-0.5, baked into Q at store time

__device__ __forceinline__ unsigned short f32_to_bf16_bits(float f) {
    union { float f; uint32_t u; } x; x.f = f;
    uint32_t u = x.u;
    if ((u & 0x7fffffffu) > 0x7f800000u) return (unsigned short)((u >> 16) | 0x0040u);
    uint32_t r = u + 0x7fffu + ((u >> 16) & 1u);   // round-to-nearest-even
    return (unsigned short)(r >> 16);
}
__device__ __forceinline__ bf16_t f32_to_bf16(float f) {
    unsigned short b = f32_to_bf16_bits(f);
    return __builtin_bit_cast(bf16_t, b);
}

// ---------------------------------------------------------------------------
// fp32 -> bf16 bulk convert
// ---------------------------------------------------------------------------
__global__ __launch_bounds__(256) void cvt_bf16_kernel(const float* __restrict__ src,
                                                       bf16_t* __restrict__ dst, int n) {
    int i = blockIdx.x * 256 + threadIdx.x;
    if (i < n) dst[i] = f32_to_bf16(src[i]);
}

// ---------------------------------------------------------------------------
// Pipelined wave GEMM helpers: A row (16xK) x 4 B subtiles -> 16x64 f32
// ---------------------------------------------------------------------------
__device__ __forceinline__ void load_a_tile(BF16x16& a, const bf16_t* arow, int k, int hi) {
    a.h[0] = *(const v8bf*)(arow + k + hi * 8);
    a.h[1] = *(const v8bf*)(arow + k + 16 + hi * 8);
}
__device__ __forceinline__ void load_b_tiles(BF16x16* bb, const bf16_t* W,
                                             int n0, int lo, int hi, int k) {
#pragma unroll
    for (int c = 0; c < 4; ++c) {
        const bf16_t* wr = W + (size_t)(n0 + c * 16 + lo) * D_SZ + k + hi * 8;
        bb[c].h[0] = *(const v8bf*)(wr);
        bb[c].h[1] = *(const v8bf*)(wr + 16);
    }
}
__device__ __forceinline__ void mma4(v8f* acc, const BF16x16& a, const BF16x16* bb) {
#pragma unroll
    for (int c = 0; c < 4; ++c)
        acc[c] = __builtin_amdgcn_wmma_f32_16x16x32_bf16(
            false, a.v, false, bb[c].v, (short)0, acc[c], false, false);
}

// ---------------------------------------------------------------------------
// QKV projection: C(4096x3072) = X(4096x1024) * Wqkv^T, wave computes 16x64.
//   n in [0,1024)    -> Q  (bf16, pre-scaled by 1/32)
//   n in [1024,2048) -> K  (f32, conv input)
//   n in [2048,3072) -> V  (f32, conv input)
// ---------------------------------------------------------------------------
__global__ __launch_bounds__(128) void qkv_gemm_kernel(const bf16_t* __restrict__ X,
                                                       const bf16_t* __restrict__ W,
                                                       bf16_t* __restrict__ Qbf,
                                                       float* __restrict__ Kf,
                                                       float* __restrict__ Vf) {
    const int lane = threadIdx.x & 31, wv = threadIdx.x >> 5;
    const int lo = lane & 15, hi = lane >> 4;
    const int wg = blockIdx.x * 4 + wv;
    const int n_idx = wg % 48, m_idx = wg / 48;       // 48 n-chunks of 64, 256 m-tiles
    const int m0 = m_idx * 16, n0 = n_idx * 64;

    v8f zero = {};
    v8f acc[4] = { zero, zero, zero, zero };
    const bf16_t* xrow = X + (size_t)(m0 + lo) * D_SZ;

    // two-deep software pipeline: issue k+32 / k+64 operands before consuming k
    BF16x16 a0, a1, b0[4], b1[4];
    load_a_tile(a0, xrow, 0, hi);
    load_b_tiles(b0, W, n0, lo, hi, 0);
#pragma unroll 1
    for (int k = 0; k < D_SZ; k += 64) {
        __builtin_prefetch(xrow + k + 256, 0, 1);
        load_a_tile(a1, xrow, k + 32, hi);
        load_b_tiles(b1, W, n0, lo, hi, k + 32);
        mma4(acc, a0, b0);
        if (k + 64 < D_SZ) {
            load_a_tile(a0, xrow, k + 64, hi);
            load_b_tiles(b0, W, n0, lo, hi, k + 64);
        }
        mma4(acc, a1, b1);
    }

    const int sec = n0 >> 10;                          // whole wave in one section
#pragma unroll
    for (int c = 0; c < 4; ++c) {
        const int col = n0 + c * 16 + lo;
        const int cl  = col & (D_SZ - 1);
#pragma unroll
        for (int r = 0; r < 8; ++r) {
            const int m = m0 + r + hi * 8;
            const float val = acc[c][r];
            if (sec == 0)      Qbf[(size_t)m * D_SZ + cl] = f32_to_bf16(val * QSCALE);
            else if (sec == 1) Kf [(size_t)m * D_SZ + cl] = val;
            else               Vf [(size_t)m * D_SZ + cl] = val;
        }
    }
}

// ---------------------------------------------------------------------------
// Grouped strided conv compression (stride 3, kernel 3, 16 groups of 64ch).
//   j==0        -> null row (from null_k / null_v)
//   1..683      -> conv output; padded pos p maps to orig t = 3*(j-1)+kk-1
//   684..JPAD-1 -> zeros
// transposed==0: Kc[(bh*JPAD + j)*64 + fo]      (key-major,  A operand for S^T)
// transposed==1: Vt[(bh*64 + fo)*JPAD + j]      (feat-major, A operand for PV)
// ---------------------------------------------------------------------------
__global__ __launch_bounds__(256) void compress_conv_kernel(const float* __restrict__ src,
                                                            const float* __restrict__ wconv,
                                                            const float* __restrict__ bconv,
                                                            const float* __restrict__ nullrow,
                                                            bf16_t* __restrict__ dst,
                                                            int transposed) {
    const int id = blockIdx.x * 256 + threadIdx.x;
    if (id >= B_SZ * JPAD * D_SZ) return;
    const int o  = id & (D_SZ - 1);
    const int j  = (id >> 10) % JPAD;
    const int bb = id / (D_SZ * JPAD);
    const int hh = o >> 6, fo = o & 63;

    float y = 0.f;
    if (j == 0) {
        y = nullrow[o];
    } else if (j <= JTOT - 1) {
        const int cp = j - 1;
        y = bconv[o];
        const float* wc = wconv + (size_t)o * 64 * 3;
#pragma unroll
        for (int kk = 0; kk < 3; ++kk) {
            const int tpos = 3 * cp + kk - 1;          // padding = 1 at front
            if (tpos >= 0) {
                const float* srow = src + ((size_t)(bb * T_SZ + tpos)) * D_SZ + hh * 64;
#pragma unroll 16
                for (int ic = 0; ic < 64; ++ic) y += srow[ic] * wc[ic * 3 + kk];
            }
        }
    }
    if (transposed) dst[((size_t)(bb * H_SZ + hh) * 64 + fo) * JPAD + j] = f32_to_bf16(y);
    else            dst[((size_t)(bb * H_SZ + hh) * JPAD + j) * 64 + fo] = f32_to_bf16(y);
}

// ---------------------------------------------------------------------------
// Flash attention per wave over one (b,h,16-query) tile, transposed form:
//   S^T = K * Q^T  (keys x queries) -> softmax column == one query == one lane
//   O^T = V^T * P^T, per-lane softmax state, single xor-16 shuffle reductions.
// Causal: key jj visible iff jj==0 || (jj<684 && qi >= 3*jj-2)
// V^T operand loads are issued before the softmax VALU section so global
// loads co-execute with the exp/shuffle work.
// ---------------------------------------------------------------------------
__global__ __launch_bounds__(128) void mca_attention_kernel(const bf16_t* __restrict__ Qbf,
                                                            const bf16_t* __restrict__ Kc,
                                                            const bf16_t* __restrict__ Vt,
                                                            bf16_t* __restrict__ Obf) {
    const int lane = threadIdx.x & 31, wv = threadIdx.x >> 5;
    const int lo = lane & 15, hi = lane >> 4;
    const int bh = blockIdx.x >> 5;                    // 32 blocks per (b,h)
    const int qt = (blockIdx.x & 31) * 4 + wv;
    const int b  = bh >> 4, h = bh & 15;
    const int i0 = qt * 16;
    const int qi = i0 + lo;

    // Q^T B operands (reused across all key blocks), Q pre-scaled by 1/32
    const bf16_t* qrow = Qbf + (size_t)(b * T_SZ + qi) * D_SZ + h * DH_SZ;
    BF16x16 bq[2];
#pragma unroll
    for (int ks = 0; ks < 2; ++ks) {
        bq[ks].h[0] = *(const v8bf*)(qrow + ks * 32 + hi * 8);
        bq[ks].h[1] = *(const v8bf*)(qrow + ks * 32 + 16 + hi * 8);
    }

    const bf16_t* kbase = Kc + (size_t)bh * JPAD * DH_SZ;
    const bf16_t* vbase = Vt + (size_t)bh * DH_SZ * JPAD;

    v8f zero = {};
    v8f acc[4] = { zero, zero, zero, zero };           // O^T, 64 feat x 16 q
    float m_run = -3.0e30f, l_run = 0.f;

    int jj_max = (i0 + 17) / 3;                        // max visible key for tile
    if (jj_max > JTOT - 1) jj_max = JTOT - 1;
    const int numj = jj_max + 1;

    for (int k0 = 0; k0 < numj; k0 += 32) {
        const bf16_t* krow0 = kbase + (size_t)(k0 + lo) * DH_SZ;
        const bf16_t* krow1 = kbase + (size_t)(k0 + 16 + lo) * DH_SZ;
        __builtin_prefetch(krow0 + 32 * DH_SZ, 0, 1);  // next key block
        __builtin_prefetch(krow1 + 32 * DH_SZ, 0, 1);

        // ---- K operands for both tiles / both k-steps, issued together
        BF16x16 a00, a01, a10, a11;
        a00.h[0] = *(const v8bf*)(krow0 + hi * 8);
        a00.h[1] = *(const v8bf*)(krow0 + 16 + hi * 8);
        a10.h[0] = *(const v8bf*)(krow1 + hi * 8);
        a10.h[1] = *(const v8bf*)(krow1 + 16 + hi * 8);
        a01.h[0] = *(const v8bf*)(krow0 + 32 + hi * 8);
        a01.h[1] = *(const v8bf*)(krow0 + 48 + hi * 8);
        a11.h[0] = *(const v8bf*)(krow1 + 32 + hi * 8);
        a11.h[1] = *(const v8bf*)(krow1 + 48 + hi * 8);

        // ---- S^T tiles: st0 = keys k0..k0+15, st1 = keys k0+16..k0+31
        v8f st0 = zero, st1 = zero;
        st0 = __builtin_amdgcn_wmma_f32_16x16x32_bf16(false, a00.v, false, bq[0].v,
                                                      (short)0, st0, false, false);
        st1 = __builtin_amdgcn_wmma_f32_16x16x32_bf16(false, a10.v, false, bq[0].v,
                                                      (short)0, st1, false, false);
        st0 = __builtin_amdgcn_wmma_f32_16x16x32_bf16(false, a01.v, false, bq[1].v,
                                                      (short)0, st0, false, false);
        st1 = __builtin_amdgcn_wmma_f32_16x16x32_bf16(false, a11.v, false, bq[1].v,
                                                      (short)0, st1, false, false);

        // ---- V^T operands are independent of the softmax: issue loads now so
        //      they complete while the VALU softmax below executes.
        BF16x16 av[4];
#pragma unroll
        for (int c = 0; c < 4; ++c) {
            const bf16_t* vrow = vbase + (size_t)(c * 16 + lo) * JPAD + k0;
            av[c].h[0] = *(const v8bf*)(vrow + hi * 16);
            av[c].h[1] = *(const v8bf*)(vrow + hi * 16 + 8);
        }

        // ---- causal + tail mask, per-lane (lane == query column)
        float p0[8], p1[8];
        float mx = -3.0e30f;
#pragma unroll
        for (int r = 0; r < 8; ++r) {
            const int jj0 = k0 + r + hi * 8;
            const int jj1 = jj0 + 16;
            const bool v0 = (jj0 == 0) || (jj0 < JTOT && qi >= 3 * jj0 - 2);
            const bool v1 = (jj1 < JTOT && qi >= 3 * jj1 - 2);
            p0[r] = v0 ? st0[r] : -3.0e30f;
            p1[r] = v1 ? st1[r] : -3.0e30f;
            mx = fmaxf(mx, fmaxf(p0[r], p1[r]));
        }
        // query q lives in lanes q and q+16: one xor-16 combine completes the row
        mx = fmaxf(mx, __shfl_xor(mx, 16, 32));
        const float m_new = fmaxf(m_run, mx);
        float ssum = 0.f;
#pragma unroll
        for (int r = 0; r < 8; ++r) {
            p0[r] = __expf(p0[r] - m_new);
            p1[r] = __expf(p1[r] - m_new);
            ssum += p0[r] + p1[r];
        }
        ssum += __shfl_xor(ssum, 16, 32);
        const float corr = __expf(m_run - m_new);
        l_run = l_run * corr + ssum;
        m_run = m_new;

        // ---- P^T (C layout) -> B operand: pack bf16 pairs, exchange halves
        unsigned d0[4], d1[4];
#pragma unroll
        for (int jd = 0; jd < 4; ++jd) {
            d0[jd] = (unsigned)f32_to_bf16_bits(p0[2 * jd]) |
                     ((unsigned)f32_to_bf16_bits(p0[2 * jd + 1]) << 16);
            d1[jd] = (unsigned)f32_to_bf16_bits(p1[2 * jd]) |
                     ((unsigned)f32_to_bf16_bits(p1[2 * jd + 1]) << 16);
        }
        BF16x16 bp;
#pragma unroll
        for (int jd = 0; jd < 4; ++jd) {
            const unsigned x0 = (unsigned)__shfl_xor((int)d0[jd], 16, 32);
            const unsigned x1 = (unsigned)__shfl_xor((int)d1[jd], 16, 32);
            bp.u[jd]     = hi ? x1 : d0[jd];           // B elems 0..7
            bp.u[4 + jd] = hi ? d1[jd] : x0;           // B elems 8..15
        }

        // ---- rescale running output and accumulate O^T += V^T * P^T
#pragma unroll
        for (int c = 0; c < 4; ++c) {
#pragma unroll
            for (int r = 0; r < 8; ++r) acc[c][r] *= corr;
            acc[c] = __builtin_amdgcn_wmma_f32_16x16x32_bf16(
                false, av[c].v, false, bp.v, (short)0, acc[c], false, false);
        }
    }

    // ---- epilogue: O^T lane holds [feat = c*16+r+hi*8][query = lo]
    const float invl = 1.0f / l_run;                   // null key always visible
    bf16_t* orow = Obf + (size_t)(b * T_SZ + i0 + lo) * D_SZ + h * DH_SZ;
#pragma unroll
    for (int c = 0; c < 4; ++c)
#pragma unroll
        for (int r = 0; r < 8; ++r)
            orow[c * 16 + r + hi * 8] = f32_to_bf16(acc[c][r] * invl);
}

// ---------------------------------------------------------------------------
// Output projection: out(4096x1024) = O(4096x1024) * Wout^T + bias  (f32 out)
// ---------------------------------------------------------------------------
__global__ __launch_bounds__(128) void out_gemm_kernel(const bf16_t* __restrict__ A,
                                                       const bf16_t* __restrict__ W,
                                                       const float* __restrict__ bias,
                                                       float* __restrict__ out) {
    const int lane = threadIdx.x & 31, wv = threadIdx.x >> 5;
    const int lo = lane & 15, hi = lane >> 4;
    const int wg = blockIdx.x * 4 + wv;
    const int n_idx = wg % 16, m_idx = wg / 16;        // 16 n-chunks of 64, 256 m-tiles
    const int m0 = m_idx * 16, n0 = n_idx * 64;

    v8f zero = {};
    v8f acc[4] = { zero, zero, zero, zero };
    const bf16_t* arow = A + (size_t)(m0 + lo) * D_SZ;

    BF16x16 a0, a1, b0[4], b1[4];
    load_a_tile(a0, arow, 0, hi);
    load_b_tiles(b0, W, n0, lo, hi, 0);
#pragma unroll 1
    for (int k = 0; k < D_SZ; k += 64) {
        __builtin_prefetch(arow + k + 256, 0, 1);
        load_a_tile(a1, arow, k + 32, hi);
        load_b_tiles(b1, W, n0, lo, hi, k + 32);
        mma4(acc, a0, b0);
        if (k + 64 < D_SZ) {
            load_a_tile(a0, arow, k + 64, hi);
            load_b_tiles(b0, W, n0, lo, hi, k + 64);
        }
        mma4(acc, a1, b1);
    }
#pragma unroll
    for (int c = 0; c < 4; ++c) {
        const int col = n0 + c * 16 + lo;
        const float bv = bias[col];
#pragma unroll
        for (int r = 0; r < 8; ++r)
            out[(size_t)(m0 + r + hi * 8) * D_SZ + col] = acc[c][r] + bv;
    }
}

// ---------------------------------------------------------------------------
extern "C" void kernel_launch(void* const* d_in, const int* in_sizes, int n_in,
                              void* d_out, int out_size, void* d_ws, size_t ws_size,
                              hipStream_t stream) {
    (void)in_sizes; (void)n_in; (void)out_size; (void)ws_size;

    const float* x      = (const float*)d_in[0];   // (2,2048,1024)
    const float* w_qkv  = (const float*)d_in[1];   // (3072,1024)
    const float* w_conv = (const float*)d_in[2];   // (1024,64,3)
    const float* b_conv = (const float*)d_in[3];   // (1024,)
    const float* null_k = (const float*)d_in[4];   // (1,1,1024)
    const float* null_v = (const float*)d_in[5];   // (1,1,1024)
    const float* w_out  = (const float*)d_in[6];   // (1024,1024)
    const float* b_out  = (const float*)d_in[7];   // (1024,)
    float* out = (float*)d_out;                    // (2,2048,1024)

    char* ws = (char*)d_ws;
    size_t off = 0;
    bf16_t* Xbf    = (bf16_t*)(ws + off); off += (size_t)4096 * 1024 * 2;   // 8 MB
    bf16_t* Wqkvbf = (bf16_t*)(ws + off); off += (size_t)3072 * 1024 * 2;   // 6 MB
    bf16_t* Woutbf = (bf16_t*)(ws + off); off += (size_t)1024 * 1024 * 2;   // 2 MB
    bf16_t* Qbf    = (bf16_t*)(ws + off); off += (size_t)4096 * 1024 * 2;   // 8 MB
    float*  Kf     = (float*)(ws + off);  off += (size_t)4096 * 1024 * 4;   // 16 MB
    float*  Vf     = (float*)(ws + off);  off += (size_t)4096 * 1024 * 4;   // 16 MB
    bf16_t* Kc     = (bf16_t*)(ws + off); off += (size_t)B_SZ * H_SZ * JPAD * 64 * 2;
    bf16_t* Vt     = (bf16_t*)(ws + off); off += (size_t)B_SZ * H_SZ * JPAD * 64 * 2;
    bf16_t* Obf    = (bf16_t*)(ws + off); off += (size_t)4096 * 1024 * 2;   // 8 MB

    // 1) fp32 -> bf16 conversions
    cvt_bf16_kernel<<<(4194304 + 255) / 256, 256, 0, stream>>>(x, Xbf, 4194304);
    cvt_bf16_kernel<<<(3145728 + 255) / 256, 256, 0, stream>>>(w_qkv, Wqkvbf, 3145728);
    cvt_bf16_kernel<<<(1048576 + 255) / 256, 256, 0, stream>>>(w_out, Woutbf, 1048576);

    // 2) QKV projection (WMMA): 256 m-tiles x 48 n-chunks = 12288 waves / 4 per block
    qkv_gemm_kernel<<<3072, 128, 0, stream>>>(Xbf, Wqkvbf, Qbf, Kf, Vf);

    // 3) conv compression: K key-major, V feature-major (transposed)
    const int conv_threads = B_SZ * JPAD * D_SZ;
    compress_conv_kernel<<<(conv_threads + 255) / 256, 256, 0, stream>>>(
        Kf, w_conv, b_conv, null_k, Kc, 0);
    compress_conv_kernel<<<(conv_threads + 255) / 256, 256, 0, stream>>>(
        Vf, w_conv, b_conv, null_v, Vt, 1);

    // 4) flash attention: 2*16 (b,h) * 32 blocks, 4 waves (q-tiles) per block
    mca_attention_kernel<<<B_SZ * H_SZ * 32, 128, 0, stream>>>(Qbf, Kc, Vt, Obf);

    // 5) output projection + bias -> d_out
    out_gemm_kernel<<<1024, 128, 0, stream>>>(Obf, Woutbf, b_out, out);
}
